// CausalSelfAttention_29472065585550
// MI455X (gfx1250) — matrix operand order repaired and checked by reading, not compiled, not run on hardware
//
#include <hip/hip_runtime.h>

typedef __bf16 bf16;
typedef __attribute__((ext_vector_type(16))) __bf16 v16bf;
typedef __attribute__((ext_vector_type(8)))  __bf16 v8bf;
typedef __attribute__((ext_vector_type(8)))  float  v8f;

#define N_EMBD 1024
#define N_HEADS 16
#define HEAD_DIM 64
#define HYPER_HEADS 4
#define RANK 8
#define LORA_SCALE 2.0f   // alpha/rank = 16/8
#define BATCH 8
#define SEQ 1024
#define MROWS (BATCH * SEQ)   // 8192

// ---------------------------------------------------------------------------
// CDNA5 async global->LDS copy (ASYNCcnt path). The LDS destination operand
// is derived from the real shared-memory pointer: on AMDGCN a generic pointer
// to LDS is {shared_aperture_hi32, lds_byte_offset_lo32}, so truncation gives
// the VDST LDS offset AND captures the pointer so the "memory" clobber makes
// the asm's LDS writes visible to alias analysis.
// ---------------------------------------------------------------------------
__device__ __forceinline__ void async_b128(void* lds_ptr, const void* gaddr) {
    unsigned off = (unsigned)(unsigned long long)(uintptr_t)lds_ptr;
    asm volatile("global_load_async_to_lds_b128 %0, %1, off"
                 :: "v"(off), "v"(gaddr) : "memory");
}
__device__ __forceinline__ void async_wait0() {
    asm volatile("s_wait_asynccnt 0x0" ::: "memory");
}

// ---------------------------------------------------------------------------
// fp32 -> bf16 flat convert
// ---------------------------------------------------------------------------
__global__ void to_bf16(const float* __restrict__ in, bf16* __restrict__ out,
                        long long n)
{
    long long i = (long long)blockIdx.x * blockDim.x + threadIdx.x;
    long long stride = (long long)gridDim.x * blockDim.x;
    for (; i < n; i += stride) out[i] = (bf16)in[i];
}

// in [R,C] fp32 -> out [C,R] bf16 (32x32 LDS tiles; R,C multiples of 32)
__global__ void __launch_bounds__(256)
transpose_to_bf16(const float* __restrict__ in, bf16* __restrict__ out,
                  int R, int C)
{
    __shared__ float tile[32][33];
    int c0 = blockIdx.x * 32, r0 = blockIdx.y * 32;
    int tx = threadIdx.x & 31, ty = threadIdx.x >> 5;   // 32 x 8
    #pragma unroll
    for (int i = 0; i < 32; i += 8)
        tile[ty + i][tx] = in[(long long)(r0 + ty + i) * C + c0 + tx];
    __syncthreads();
    #pragma unroll
    for (int i = 0; i < 32; i += 8)
        out[(long long)(c0 + ty + i) * R + r0 + tx] = (bf16)tile[tx][ty + i];
}

// ---------------------------------------------------------------------------
// Big GEMM: C[M,N] = A[M,K] * Wt^T + bias. A bf16 [M,K], Wt bf16 [N,K].
// 256 threads = 8 waves, block tile 128x128, wave tile 32x64, K step 32.
// Double-buffered LDS filled via global_load_async_to_lds_b128.
// All fragments loaded before the WMMA chain -> 8 back-to-back WMMAs.
// ---------------------------------------------------------------------------
template <int C_BF16>
__global__ void __launch_bounds__(256)
gemm_big(const bf16* __restrict__ A, int lda,
         const bf16* __restrict__ Wt, int ldw,
         const float* __restrict__ bias,
         float* __restrict__ Cf, bf16* __restrict__ Cb, int ldc, int K)
{
    const int nBase = blockIdx.x * 128;
    const int mBase = blockIdx.y * 128;
    const int tid   = threadIdx.x;
    const int lane  = tid & 31;
    const int wave  = tid >> 5;
    const int wm    = wave >> 1;    // 0..3  -> rows [32*wm, +32)
    const int wn    = wave & 1;     // 0..1  -> cols [64*wn, +64)
    const int hl    = (lane >> 4) & 1;
    const int l15   = lane & 15;

    // LDS byte layout: rows of 64B data padded to 80B stride
    __shared__ alignas(16) char smem[4 * 128 * 80];
    bf16* sb = (bf16*)smem;
    const unsigned ASB[2] = { 0u, 10240u };
    const unsigned BSB[2] = { 20480u, 30720u };

    v8f acc[2][4];
    for (int ms = 0; ms < 2; ++ms)
        for (int a = 0; a < 4; ++a)
            for (int i = 0; i < 8; ++i) acc[ms][a][i] = 0.0f;

    auto stage = [&](int k0, int buf) {
        #pragma unroll
        for (int i = 0; i < 2; ++i) {
            int cid = i * 256 + tid;          // 512 chunks of 16B per tile
            int r = cid >> 2, j = cid & 3;
            async_b128(smem + ASB[buf] + r * 80 + j * 16,
                       A + (long long)(mBase + r) * lda + k0 + j * 8);
            async_b128(smem + BSB[buf] + r * 80 + j * 16,
                       Wt + (long long)(nBase + r) * ldw + k0 + j * 8);
        }
    };

    const int ksteps = K >> 5;
    stage(0, 0);
    for (int kt = 0; kt < ksteps; ++kt) {
        async_wait0();
        __syncthreads();
        if (kt + 1 < ksteps) stage((kt + 1) * 32, (kt + 1) & 1);

        const unsigned asE = ASB[kt & 1] >> 1;   // bf16 element bases
        const unsigned bsE = BSB[kt & 1] >> 1;

        // load ALL fragments first
        v16bf af[2], bfr[4];
        #pragma unroll
        for (int ms = 0; ms < 2; ++ms) {
            int m = wm * 32 + ms * 16 + l15;
            #pragma unroll
            for (int v = 0; v < 8; ++v) {
                int kk = ((v < 4) ? 0 : 16) + (hl ? 8 : 0) + 2 * (v & 3);
                af[ms][2 * v]     = sb[asE + m * 40 + kk];
                af[ms][2 * v + 1] = sb[asE + m * 40 + kk + 1];
            }
        }
        #pragma unroll
        for (int a = 0; a < 4; ++a) {
            int n = wn * 64 + a * 16 + l15;
            #pragma unroll
            for (int v = 0; v < 8; ++v) {
                int kk = ((v < 4) ? 0 : 16) + (hl ? 8 : 0) + 2 * (v & 3);
                bfr[a][2 * v]     = sb[bsE + n * 40 + kk];
                bfr[a][2 * v + 1] = sb[bsE + n * 40 + kk + 1];
            }
        }
        // 8 back-to-back WMMAs (no D->A/B hazards between them)
        #pragma unroll
        for (int a = 0; a < 4; ++a)
            #pragma unroll
            for (int ms = 0; ms < 2; ++ms)
                acc[ms][a] = __builtin_amdgcn_wmma_f32_16x16x32_bf16(
                    false, af[ms], false, bfr[a], (short)0, acc[ms][a], false, false);
    }

    #pragma unroll
    for (int ms = 0; ms < 2; ++ms) {
        #pragma unroll
        for (int a = 0; a < 4; ++a) {
            int col = nBase + wn * 64 + a * 16 + l15;
            float bv = bias ? bias[col] : 0.0f;
            #pragma unroll
            for (int v = 0; v < 8; ++v) {
                int row = mBase + wm * 32 + ms * 16 + v + (hl ? 8 : 0);
                float val = acc[ms][a][v] + bv;
                if (C_BF16) Cb[(long long)row * ldc + col] = (bf16)val;
                else        Cf[(long long)row * ldc + col] = val;
            }
        }
    }
}

// ---------------------------------------------------------------------------
// Small batched GEMM (hypernet projections): 64x64 tile, 128 threads,
// async double-buffered staging. A bf16 [M,K], W bf16 [N,K], bf16 out.
// ---------------------------------------------------------------------------
__global__ void __launch_bounds__(128)
gemm_small(const bf16* __restrict__ A, long long aBatch, int lda,
           const bf16* __restrict__ W, long long wBatch, int ldw,
           const float* __restrict__ bias, long long biasBatch,
           bf16* __restrict__ C, long long cBatch, int ldc, int K)
{
    const int batch = blockIdx.z;
    A += (long long)batch * aBatch;
    W += (long long)batch * wBatch;
    C += (long long)batch * cBatch;
    const float* bi = bias + (long long)batch * biasBatch;

    const int nBase = blockIdx.x * 64;
    const int mBase = blockIdx.y * 64;
    const int tid   = threadIdx.x;
    const int lane  = tid & 31;
    const int wave  = tid >> 5;
    const int hl    = (lane >> 4) & 1;
    const int l15   = lane & 15;

    __shared__ alignas(16) char smem[4 * 64 * 80];
    bf16* sb = (bf16*)smem;
    const unsigned ASB[2] = { 0u, 5120u };
    const unsigned BSB[2] = { 10240u, 15360u };

    v8f acc[4];
    for (int a = 0; a < 4; ++a)
        for (int i = 0; i < 8; ++i) acc[a][i] = 0.0f;

    auto stage = [&](int k0, int buf) {
        #pragma unroll
        for (int i = 0; i < 2; ++i) {
            int cid = i * 128 + tid;          // 256 chunks per tile
            int r = cid >> 2, j = cid & 3;
            async_b128(smem + ASB[buf] + r * 80 + j * 16,
                       A + (long long)(mBase + r) * lda + k0 + j * 8);
            async_b128(smem + BSB[buf] + r * 80 + j * 16,
                       W + (long long)(nBase + r) * ldw + k0 + j * 8);
        }
    };

    const int ksteps = K >> 5;
    stage(0, 0);
    for (int kt = 0; kt < ksteps; ++kt) {
        async_wait0();
        __syncthreads();
        if (kt + 1 < ksteps) stage((kt + 1) * 32, (kt + 1) & 1);

        const unsigned asE = ASB[kt & 1] >> 1;
        const unsigned bsE = BSB[kt & 1] >> 1;

        v16bf af, bfr[4];
        int m = wave * 16 + l15;
        #pragma unroll
        for (int v = 0; v < 8; ++v) {
            int kk = ((v < 4) ? 0 : 16) + (hl ? 8 : 0) + 2 * (v & 3);
            af[2 * v]     = sb[asE + m * 40 + kk];
            af[2 * v + 1] = sb[asE + m * 40 + kk + 1];
        }
        #pragma unroll
        for (int a = 0; a < 4; ++a) {
            int n = a * 16 + l15;
            #pragma unroll
            for (int v = 0; v < 8; ++v) {
                int kk = ((v < 4) ? 0 : 16) + (hl ? 8 : 0) + 2 * (v & 3);
                bfr[a][2 * v]     = sb[bsE + n * 40 + kk];
                bfr[a][2 * v + 1] = sb[bsE + n * 40 + kk + 1];
            }
        }
        #pragma unroll
        for (int a = 0; a < 4; ++a)
            acc[a] = __builtin_amdgcn_wmma_f32_16x16x32_bf16(
                false, af, false, bfr[a], (short)0, acc[a], false, false);
    }

    #pragma unroll
    for (int a = 0; a < 4; ++a) {
        int col = nBase + a * 16 + l15;
        float bv = bi[col];
        #pragma unroll
        for (int v = 0; v < 8; ++v) {
            int row = mBase + wave * 16 + v + (hl ? 8 : 0);
            C[(long long)row * ldc + col] = (bf16)(acc[a][v] + bv);
        }
    }
}

// ---------------------------------------------------------------------------
// Main causal flash attention. 128 threads = 4 waves, one 64-query tile of
// one (b,h). Async-staged Q/K/V (K/V double-buffered), WMMA bf16 core,
// online softmax in C-layout registers. Q fragments hoisted out of the
// K-loop (Q LDS tile is loop-invariant). 1/sqrt(64) applied post-WMMA.
// qkv bf16 [B*T, 3072]; vT bf16 [B,H, D, T]; y bf16 [B*T, 1024] head-major.
// ---------------------------------------------------------------------------
__global__ void __launch_bounds__(128)
flash_attn(const bf16* __restrict__ qkv, const bf16* __restrict__ vT,
           bf16* __restrict__ y)
{
    const int qt   = blockIdx.x;   // 16 tiles of 64 queries
    const int h    = blockIdx.y;
    const int b    = blockIdx.z;
    const int tid  = threadIdx.x;
    const int lane = tid & 31;
    const int wave = tid >> 5;
    const int hl   = (lane >> 4) & 1;
    const int l15  = lane & 15;

    // rows of 128B (64 bf16) padded to 144B stride
    __shared__ alignas(16) char smem[6 * 64 * 144];   // Q | K0 K1 | V0 V1 | P
    bf16* sb = (bf16*)smem;
    const unsigned QSB    = 0u;
    const unsigned KSB[2] = {  9216u, 18432u };
    const unsigned VSB[2] = { 27648u, 36864u };
    const unsigned PSB    = 46080u;

    const long long rs = 3 * N_EMBD;
    const bf16* qbase = qkv + (long long)(b * SEQ) * rs + h * HEAD_DIM;
    const bf16* kbase = qbase + N_EMBD;
    const bf16* vbase = vT + (long long)(b * N_HEADS + h) * HEAD_DIM * SEQ;

    // stage Q tile (64 rows x 128B = 512 16B-chunks)
    #pragma unroll
    for (int i = 0; i < 4; ++i) {
        int cid = i * 128 + tid;
        int r = cid >> 3, j = cid & 7;
        async_b128(smem + QSB + r * 144 + j * 16,
                   qbase + (long long)(qt * 64 + r) * rs + j * 8);
    }
    auto stageKV = [&](int kt, int buf) {
        #pragma unroll
        for (int i = 0; i < 4; ++i) {
            int cid = i * 128 + tid;
            int r = cid >> 3, j = cid & 7;
            async_b128(smem + KSB[buf] + r * 144 + j * 16,
                       kbase + (long long)(kt * 64 + r) * rs + j * 8);
            async_b128(smem + VSB[buf] + r * 144 + j * 16,
                       vbase + (long long)r * SEQ + kt * 64 + j * 8);  // r = d
        }
    };
    stageKV(0, 0);

    async_wait0();
    __syncthreads();

    // Q A-fragments for both WMMA K-steps: loop-invariant, load once
    const unsigned qsE = QSB >> 1;
    const unsigned psE = PSB >> 1;
    v16bf qaf[2];
    #pragma unroll
    for (int ks = 0; ks < 2; ++ks) {
        int m = wave * 16 + l15;
        #pragma unroll
        for (int v = 0; v < 8; ++v) {
            int kk = ks * 32 + ((v < 4) ? 0 : 16) + (hl ? 8 : 0) + 2 * (v & 3);
            qaf[ks][2 * v]     = sb[qsE + m * 72 + kk];
            qaf[ks][2 * v + 1] = sb[qsE + m * 72 + kk + 1];
        }
    }

    float rm[8], rl[8];
    #pragma unroll
    for (int v = 0; v < 8; ++v) { rm[v] = -3.0e38f; rl[v] = 0.0f; }
    v8f o[4];
    for (int a = 0; a < 4; ++a)
        for (int i = 0; i < 8; ++i) o[a][i] = 0.0f;

    for (int kt = 0; kt <= qt; ++kt) {
        if (kt > 0) {
            async_wait0();
            __syncthreads();
        }
        if (kt < qt) stageKV(kt + 1, (kt + 1) & 1);

        const unsigned ksE = KSB[kt & 1] >> 1;
        const unsigned vsE = VSB[kt & 1] >> 1;

        // S = Q * K^T (inner dim d = 64, two WMMA K-steps)
        v8f s[4];
        for (int a = 0; a < 4; ++a)
            for (int i = 0; i < 8; ++i) s[a][i] = 0.0f;
        #pragma unroll
        for (int ks = 0; ks < 2; ++ks) {
            v16bf kb[4];
            #pragma unroll
            for (int a = 0; a < 4; ++a) {
                int n = a * 16 + l15;
                #pragma unroll
                for (int v = 0; v < 8; ++v) {
                    int kk = ks * 32 + ((v < 4) ? 0 : 16) + (hl ? 8 : 0) + 2 * (v & 3);
                    kb[a][2 * v]     = sb[ksE + n * 72 + kk];
                    kb[a][2 * v + 1] = sb[ksE + n * 72 + kk + 1];
                }
            }
            #pragma unroll
            for (int a = 0; a < 4; ++a)
                s[a] = __builtin_amdgcn_wmma_f32_16x16x32_bf16(
                    false, qaf[ks], false, kb[a], (short)0, s[a], false, false);
        }

        // scale + causal mask + online softmax (16-lane-half row reductions)
        #pragma unroll
        for (int v = 0; v < 8; ++v) {
            int qg = qt * 64 + wave * 16 + v + (hl ? 8 : 0);
            float lm = -3.0e38f;
            #pragma unroll
            for (int a = 0; a < 4; ++a) {
                int kg = kt * 64 + a * 16 + l15;
                float sv = s[a][v] * 0.125f;
                if (kg > qg) sv = -3.0e38f;
                s[a][v] = sv;
                lm = fmaxf(lm, sv);
            }
            for (int off = 1; off < 16; off <<= 1)
                lm = fmaxf(lm, __shfl_xor(lm, off, 32));
            float mn   = fmaxf(rm[v], lm);
            float corr = __expf(rm[v] - mn);
            rm[v] = mn;
            float ps = 0.0f;
            #pragma unroll
            for (int a = 0; a < 4; ++a) {
                float p = __expf(s[a][v] - mn);
                s[a][v] = p;
                ps += p;
            }
            for (int off = 1; off < 16; off <<= 1)
                ps += __shfl_xor(ps, off, 32);
            rl[v] = rl[v] * corr + ps;
            #pragma unroll
            for (int a = 0; a < 4; ++a) o[a][v] *= corr;
        }

        // C-layout -> A-layout via LDS
        #pragma unroll
        for (int a = 0; a < 4; ++a)
            #pragma unroll
            for (int v = 0; v < 8; ++v)
                sb[psE + (wave * 16 + v + (hl ? 8 : 0)) * 72 + a * 16 + l15] =
                    (bf16)s[a][v];
        __syncthreads();

        // O += P * V (inner dim = 64 keys)
        #pragma unroll
        for (int ks = 0; ks < 2; ++ks) {
            v16bf paf, vb[4];
            int m = wave * 16 + l15;
            #pragma unroll
            for (int v = 0; v < 8; ++v) {
                int kk = ks * 32 + ((v < 4) ? 0 : 16) + (hl ? 8 : 0) + 2 * (v & 3);
                paf[2 * v]     = sb[psE + m * 72 + kk];
                paf[2 * v + 1] = sb[psE + m * 72 + kk + 1];
            }
            #pragma unroll
            for (int a = 0; a < 4; ++a) {
                int n = a * 16 + l15;    // d column
                #pragma unroll
                for (int v = 0; v < 8; ++v) {
                    int kk = ks * 32 + ((v < 4) ? 0 : 16) + (hl ? 8 : 0) + 2 * (v & 3);
                    vb[a][2 * v]     = sb[vsE + n * 72 + kk];
                    vb[a][2 * v + 1] = sb[vsE + n * 72 + kk + 1];
                }
            }
            #pragma unroll
            for (int a = 0; a < 4; ++a)
                o[a] = __builtin_amdgcn_wmma_f32_16x16x32_bf16(
                    false, paf, false, vb[a], (short)0, o[a], false, false);
        }
    }

    #pragma unroll
    for (int a = 0; a < 4; ++a) {
        #pragma unroll
        for (int v = 0; v < 8; ++v) {
            int t = qt * 64 + wave * 16 + v + (hl ? 8 : 0);
            int d = a * 16 + l15;
            y[((long long)(b * SEQ) + t) * N_EMBD + h * HEAD_DIM + d] =
                (bf16)(o[a][v] * (1.0f / rl[v]));
        }
    }
}

// ---------------------------------------------------------------------------
// Hypernet causal MHA (4 heads, dim 16): scalar online softmax, LDS tiles.
// hqkv bf16 [H][B*T][192]; ah bf16 [H][B*T][64].
// ---------------------------------------------------------------------------
__global__ void __launch_bounds__(128)
hyper_attn(const bf16* __restrict__ hqkv, bf16* __restrict__ ah)
{
    const int qt  = blockIdx.x;
    const int b   = blockIdx.y >> 2;
    const int hh  = blockIdx.y & 3;
    const int h   = blockIdx.z;
    const int tid = threadIdx.x;

    const bf16* base = hqkv + (long long)h * MROWS * (3 * HEAD_DIM)
                            + (long long)(b * SEQ) * (3 * HEAD_DIM);

    __shared__ float Kl[128][17];
    __shared__ float Vl[128][17];

    const int q = qt * 128 + tid;
    float qv[16];
    #pragma unroll
    for (int d = 0; d < 16; ++d)
        qv[d] = (float)base[(long long)q * (3 * HEAD_DIM) + hh * 16 + d] * 0.25f;

    float m = -3.0e38f, l = 0.0f, o[16];
    #pragma unroll
    for (int d = 0; d < 16; ++d) o[d] = 0.0f;

    for (int kt = 0; kt <= qt; ++kt) {
        __syncthreads();
        // stage K/V as 16B bf16x8 vector loads, convert to fp32 in LDS
        #pragma unroll
        for (int i = 0; i < 2; ++i) {
            int cid = i * 128 + tid;          // 256 chunks (128 rows x 2)
            int r = cid >> 1, hf = cid & 1;
            long long row = (long long)(kt * 128 + r) * (3 * HEAD_DIM);
            v8bf kv = *(const v8bf*)(base + row + 64  + hh * 16 + hf * 8);
            v8bf vv = *(const v8bf*)(base + row + 128 + hh * 16 + hf * 8);
            #pragma unroll
            for (int e = 0; e < 8; ++e) {
                Kl[r][hf * 8 + e] = (float)kv[e];
                Vl[r][hf * 8 + e] = (float)vv[e];
            }
        }
        __syncthreads();
        int jmax = (kt == qt) ? (tid + 1) : 128;
        for (int j = 0; j < jmax; ++j) {
            float s = 0.0f;
            #pragma unroll
            for (int d = 0; d < 16; ++d) s += qv[d] * Kl[j][d];
            if (s > m) {
                float corr = __expf(m - s);
                l *= corr;
                #pragma unroll
                for (int d = 0; d < 16; ++d) o[d] *= corr;
                m = s;
            }
            float p = __expf(s - m);
            l += p;
            #pragma unroll
            for (int d = 0; d < 16; ++d) o[d] += p * Vl[j][d];
        }
    }
    float inv = 1.0f / l;
    bf16* op = ah + (long long)h * MROWS * HEAD_DIM
                  + ((long long)(b * SEQ) + q) * HEAD_DIM + hh * 16;
    #pragma unroll
    for (int d = 0; d < 16; ++d) op[d] = (bf16)(o[d] * inv);
}

// ctx[b,h,:] = mean over T of hidden[h][b*T + t][:]   (bf16 in, fp32 out)
__global__ void ctx_mean(const bf16* __restrict__ hid, float* __restrict__ ctx)
{
    int bh = blockIdx.x;
    int b = bh / N_HEADS, h = bh % N_HEADS;
    int d = threadIdx.x;
    const bf16* p = hid + (long long)h * MROWS * HEAD_DIM
                        + (long long)(b * SEQ) * HEAD_DIM + d;
    float s = 0.0f;
    for (int t = 0; t < SEQ; ++t) s += (float)p[(long long)t * HEAD_DIM];
    ctx[(long long)bh * HEAD_DIM + d] = s * (1.0f / SEQ);
}

// Af[bh,i] = ctx[bh] . A_w[h,i,:], Bf likewise (i in [0,512))
__global__ void __launch_bounds__(512)
lora_factors(const float* __restrict__ ctx, const float* __restrict__ Aw,
             const float* __restrict__ Bw, float* __restrict__ Af,
             float* __restrict__ Bf)
{
    int bh = blockIdx.x;
    int h = bh % N_HEADS;
    __shared__ float cl[64];
    if (threadIdx.x < 64) cl[threadIdx.x] = ctx[(long long)bh * 64 + threadIdx.x];
    __syncthreads();
    int i = threadIdx.x;
    const float* aw = Aw + ((long long)h * 512 + i) * 64;
    const float* bw = Bw + ((long long)h * 512 + i) * 64;
    float sa = 0.0f, sb = 0.0f;
    for (int k = 0; k < 64; ++k) { sa += cl[k] * aw[k]; sb += cl[k] * bw[k]; }
    Af[(long long)bh * 512 + i] = sa;
    Bf[(long long)bh * 512 + i] = sb;
}

// v_new = v + SCALE*(v@A)@Bm. Reads bf16 v from qkv, writes TRANSPOSED
// bf16 vT[b,h,d,t] so flash attention V tiles are contiguous row copies.
__global__ void __launch_bounds__(128)
apply_lora(const bf16* __restrict__ qkv, const float* __restrict__ Af,
           const float* __restrict__ Bf, bf16* __restrict__ vT)
{
    int bh = blockIdx.y;
    int b = bh / N_HEADS, h = bh % N_HEADS;
    __shared__ float Al[512], Bl[512];
    for (int i = threadIdx.x; i < 512; i += 128) {
        Al[i] = Af[(long long)bh * 512 + i];
        Bl[i] = Bf[(long long)bh * 512 + i];
    }
    __syncthreads();
    int t = blockIdx.x * 128 + threadIdx.x;
    const bf16* vp = qkv + ((long long)(b * SEQ) + t) * (3 * N_EMBD)
                         + 2 * N_EMBD + h * HEAD_DIM;
    float vv[64];
    #pragma unroll
    for (int d = 0; d < 64; ++d) vv[d] = (float)vp[d];
    float tmp[RANK];
    #pragma unroll
    for (int r = 0; r < RANK; ++r) {
        float s = 0.0f;
        #pragma unroll
        for (int d = 0; d < 64; ++d) s += vv[d] * Al[d * RANK + r];
        tmp[r] = s;
    }
    #pragma unroll
    for (int d = 0; d < 64; ++d) {
        float s = 0.0f;
        #pragma unroll
        for (int r = 0; r < RANK; ++r) s += tmp[r] * Bl[r * 64 + d];
        vT[((long long)bh * HEAD_DIM + d) * SEQ + t] =
            (bf16)(vv[d] + LORA_SCALE * s);
    }
}

// ---------------------------------------------------------------------------
extern "C" void kernel_launch(void* const* d_in, const int* in_sizes, int n_in,
                              void* d_out, int out_size, void* d_ws, size_t ws_size,
                              hipStream_t stream)
{
    const float* x        = (const float*)d_in[0];
    const float* W_attn   = (const float*)d_in[1];
    const float* b_attn   = (const float*)d_in[2];
    const float* W_proj   = (const float*)d_in[3];
    const float* b_proj   = (const float*)d_in[4];
    const float* hy_in_w  = (const float*)d_in[5];
    const float* hy_in_b  = (const float*)d_in[6];
    const float* hy_out_w = (const float*)d_in[7];
    const float* hy_out_b = (const float*)d_in[8];
    const float* hy_A_w   = (const float*)d_in[9];
    const float* hy_B_w   = (const float*)d_in[10];
    float* out = (float*)d_out;

    char* ws = (char*)d_ws;
    size_t off = 0;
    auto alloc = [&](size_t bytes) {
        char* p = ws + off;
        off = (off + bytes + 255) & ~(size_t)255;
        return p;
    };
    bf16*  xbf    = (bf16*)alloc((size_t)MROWS * N_EMBD * 2);
    bf16*  WattnT = (bf16*)alloc((size_t)3 * N_EMBD * N_EMBD * 2);
    bf16*  WprojT = (bf16*)alloc((size_t)N_EMBD * N_EMBD * 2);
    bf16*  hyiw   = (bf16*)alloc((size_t)2 * N_HEADS * 192 * 64 * 2);
    bf16*  hyow   = (bf16*)alloc((size_t)2 * N_HEADS * 64 * 64 * 2);
    bf16*  qkv    = (bf16*)alloc((size_t)MROWS * 3 * N_EMBD * 2);
    bf16*  hqkv   = (bf16*)alloc((size_t)N_HEADS * MROWS * 192 * 2);
    bf16*  ahb    = (bf16*)alloc((size_t)N_HEADS * MROWS * 64 * 2);
    bf16*  hid0   = (bf16*)alloc((size_t)N_HEADS * MROWS * 64 * 2);
    bf16*  hid1   = (bf16*)alloc((size_t)N_HEADS * MROWS * 64 * 2);
    float* ctxb   = (float*)alloc((size_t)BATCH * N_HEADS * 64 * 4);
    float* Af     = (float*)alloc((size_t)BATCH * N_HEADS * 512 * 4);
    float* Bf     = (float*)alloc((size_t)BATCH * N_HEADS * 512 * 4);
    bf16*  vT     = (bf16*)alloc((size_t)BATCH * N_HEADS * 64 * SEQ * 2);
    bf16*  ybuf   = (bf16*)alloc((size_t)MROWS * N_EMBD * 2);

    // 0) precision/layout conversions
    to_bf16<<<2048, 256, 0, stream>>>(x, xbf, (long long)MROWS * N_EMBD);
    to_bf16<<<512, 256, 0, stream>>>(hy_in_w, hyiw, (long long)2 * N_HEADS * 192 * 64);
    to_bf16<<<256, 256, 0, stream>>>(hy_out_w, hyow, (long long)2 * N_HEADS * 64 * 64);
    transpose_to_bf16<<<dim3(3 * N_EMBD / 32, N_EMBD / 32), 256, 0, stream>>>(
        W_attn, WattnT, N_EMBD, 3 * N_EMBD);
    transpose_to_bf16<<<dim3(N_EMBD / 32, N_EMBD / 32), 256, 0, stream>>>(
        W_proj, WprojT, N_EMBD, N_EMBD);

    // 1) QKV projection -> bf16 qkv
    gemm_big<1><<<dim3(3 * N_EMBD / 128, MROWS / 128), 256, 0, stream>>>(
        xbf, N_EMBD, WattnT, N_EMBD, b_attn, nullptr, qkv, 3 * N_EMBD, N_EMBD);

    // 2) hypernet: 2 layers, batched over 16 heads
    for (int l = 0; l < 2; ++l) {
        const bf16* Ain = (l == 0) ? xbf : hid0;
        long long aStride = (l == 0) ? HEAD_DIM : (long long)MROWS * HEAD_DIM;
        int lda = (l == 0) ? N_EMBD : HEAD_DIM;
        gemm_small<<<dim3(192 / 64, MROWS / 64, N_HEADS), 128, 0, stream>>>(
            Ain, aStride, lda,
            hyiw + (long long)l * N_HEADS * 192 * 64, (long long)192 * 64, 64,
            hy_in_b + (long long)l * N_HEADS * 192, 192,
            hqkv, (long long)MROWS * 192, 192, 64);
        hyper_attn<<<dim3(SEQ / 128, BATCH * HYPER_HEADS, N_HEADS), 128, 0, stream>>>(
            hqkv, ahb);
        bf16* hOut = (l == 0) ? hid0 : hid1;
        gemm_small<<<dim3(1, MROWS / 64, N_HEADS), 128, 0, stream>>>(
            ahb, (long long)MROWS * 64, 64,
            hyow + (long long)l * N_HEADS * 64 * 64, (long long)64 * 64, 64,
            hy_out_b + (long long)l * N_HEADS * 64, 64,
            hOut, (long long)MROWS * 64, 64, 64);
    }

    // 3) ctx mean, LoRA factors, apply delta to V (writes transposed V)
    ctx_mean<<<BATCH * N_HEADS, HEAD_DIM, 0, stream>>>(hid1, ctxb);
    lora_factors<<<BATCH * N_HEADS, 512, 0, stream>>>(ctxb, hy_A_w, hy_B_w, Af, Bf);
    apply_lora<<<dim3(SEQ / 128, BATCH * N_HEADS), 128, 0, stream>>>(qkv, Af, Bf, vT);

    // 4) main causal SDPA (async-staged WMMA flash attention)
    flash_attn<<<dim3(SEQ / 64, N_HEADS, BATCH), 128, 0, stream>>>(qkv, vT, ybuf);

    // 5) output projection -> fp32 d_out
    gemm_big<0><<<dim3(N_EMBD / 128, MROWS / 128), 256, 0, stream>>>(
        ybuf, N_EMBD, WprojT, N_EMBD, b_proj, out, nullptr, N_EMBD, N_EMBD);

    (void)in_sizes; (void)n_in; (void)out_size; (void)ws_size;
}
